// GCNNetAST_9320079033238
// MI455X (gfx1250) — compile-verified, matrix operand order checked by reading, not run on hardware
//
#include <hip/hip_runtime.h>
#include <hip/hip_bf16.h>

typedef __attribute__((ext_vector_type(16))) __bf16 v16bf;
typedef __attribute__((ext_vector_type(8)))  __bf16 v8bf;
typedef __attribute__((ext_vector_type(8)))  float  v8f;

#define LRELU(x) ((x) >= 0.0f ? (x) : 0.01f * (x))

// ---------------------------------------------------------------------------
// small elementwise / setup kernels
// ---------------------------------------------------------------------------

// h0_bf[i, k] = k < D ? bf16(emb[tok[i]*D + k]) : 0     (KP padded to mult of 32)
__global__ __launch_bounds__(256)
void embed_pad_kernel(const int* __restrict__ tok, const float* __restrict__ emb,
                      __bf16* __restrict__ h0, int N, int D, int KP) {
    size_t idx = (size_t)blockIdx.x * blockDim.x + threadIdx.x;
    size_t total = (size_t)N * KP;
    if (idx >= total) return;
    int i = (int)(idx / KP), k = (int)(idx % KP);
    float v = (k < D) ? emb[(size_t)tok[i] * D + k] : 0.0f;
    h0[idx] = (__bf16)v;
}

// TRANSPOSED cast: Wt[j, k] = (k < K && j < Nt) ? bf16(W[k*Nt + j]) : 0, over [NP, KP]
__global__ __launch_bounds__(256)
void wcast_padT_kernel(const float* __restrict__ W, __bf16* __restrict__ Wt,
                       int K, int KP, int Nt, int NP) {
    size_t idx = (size_t)blockIdx.x * blockDim.x + threadIdx.x;
    size_t total = (size_t)NP * KP;
    if (idx >= total) return;
    int j = (int)(idx / KP), k = (int)(idx % KP);
    float v = (k < K && j < Nt) ? W[(size_t)k * Nt + j] : 0.0f;
    Wt[idx] = (__bf16)v;
}

__global__ __launch_bounds__(256)
void deg_init_kernel(float* __restrict__ deg, int N) {
    int i = blockIdx.x * blockDim.x + threadIdx.x;
    if (i < N) deg[i] = 1.0f;   // self loop
}

__global__ __launch_bounds__(256)
void deg_edges_kernel(const int* __restrict__ dst, float* __restrict__ deg, int E) {
    int e = blockIdx.x * blockDim.x + threadIdx.x;
    if (e < E) atomicAdd(&deg[dst[e]], 1.0f);
}

__global__ __launch_bounds__(256)
void dinv_kernel(float* __restrict__ deg, int N) {
    int i = blockIdx.x * blockDim.x + threadIdx.x;
    if (i < N) { float d = deg[i]; deg[i] = (d > 0.0f) ? rsqrtf(d) : 0.0f; }
}

// ---------------------------------------------------------------------------
// bf16 WMMA GEMM: C_bf16[m,n] = dinv[m] * sum_k A[m,k]*Bt[n,k]
// A:  [M, KP] bf16 row-major (KP%32==0, M%64==0)
// Bt: [NP, KP] bf16 row-major (pre-transposed weights, zero padded, NP%256==0)
// Block = 256 threads = 8 waves (2x4); block tile 64x256; wave tile 32x64
// => 8 v_wmma_f32_16x16x32_bf16 per wave per K-step, LDS double-buffered.
// ---------------------------------------------------------------------------
#define TM 64
#define TNB 256
#define TK 32

static __device__ inline v16bf frag16(const __bf16* rowptr, int ks) {
    // halves at K = ks..ks+7 and K = ks+16..ks+23 (ISA 16-bit operand layout)
    v8bf lo = *(const v8bf*)(rowptr + ks);
    v8bf hi = *(const v8bf*)(rowptr + ks + 16);
    return __builtin_shufflevector(lo, hi, 0,1,2,3,4,5,6,7,8,9,10,11,12,13,14,15);
}

__global__ __launch_bounds__(256)
void gemm_bf16_rowscale(const __bf16* __restrict__ A, const __bf16* __restrict__ Bt,
                        const float* __restrict__ dinv, __bf16* __restrict__ Cout,
                        int M, int KP, int Nt) {
    __shared__ __bf16 As[2][TM][TK];    //  8 KB  (double buffered)
    __shared__ __bf16 Bs[2][TNB][TK];   // 32 KB  (double buffered, row = out col)

    const int tid  = threadIdx.x;
    const int lane = tid & 31;
    const int wave = tid >> 5;        // 0..7
    const int wm   = wave >> 2;       // 0..1 : 32-row slab
    const int wn   = wave & 3;        // 0..3 : 64-col slab
    const int m0   = blockIdx.y * TM;
    const int n0   = blockIdx.x * TNB;

    // cooperative load mappings (all contiguous 16B)
    const int a_row = tid >> 2;           // 0..63
    const int a_col = (tid & 3) * 8;      // 8 contiguous bf16

    const int r  = lane & 15;             // fragment row/col within 16
    const int ks = (lane >> 4) * 8;       // K half-offset

    const __bf16* agBase = A  + (size_t)(m0 + a_row) * KP + a_col;
    const __bf16* bgBase = Bt + (size_t)(n0 + tid) * KP;

    v8f acc[2][4] = {};

    // ---- stage first tile into buffer 0 ----
    *(v8bf*)&As[0][a_row][a_col] = *(const v8bf*)agBase;
    #pragma unroll
    for (int j = 0; j < 4; ++j)
        *(v8bf*)&Bs[0][tid][j * 8] = *(const v8bf*)(bgBase + j * 8);
    __syncthreads();

    for (int k0 = 0; k0 < KP; k0 += TK) {
        const int cur = (k0 / TK) & 1;
        const int nk  = k0 + TK;
        if (nk < KP) {
            const int nxt = cur ^ 1;
            *(v8bf*)&As[nxt][a_row][a_col] = *(const v8bf*)(agBase + nk);
            #pragma unroll
            for (int j = 0; j < 4; ++j)
                *(v8bf*)&Bs[nxt][tid][j * 8] = *(const v8bf*)(bgBase + nk + j * 8);
            if (nk + TK < KP) {
                __builtin_prefetch(agBase + nk + TK, 0, 3);
                __builtin_prefetch(bgBase + nk + TK, 0, 3);
            }
        }

        v16bf af[2], bb[4];
        #pragma unroll
        for (int ai = 0; ai < 2; ++ai) af[ai] = frag16(&As[cur][wm * 32 + ai * 16 + r][0], ks);
        #pragma unroll
        for (int bi = 0; bi < 4; ++bi) bb[bi] = frag16(&Bs[cur][wn * 64 + bi * 16 + r][0], ks);

        #pragma unroll
        for (int ai = 0; ai < 2; ++ai)
            #pragma unroll
            for (int bi = 0; bi < 4; ++bi)
                acc[ai][bi] = __builtin_amdgcn_wmma_f32_16x16x32_bf16(
                    false, af[ai], false, bb[bi], (short)0, acc[ai][bi], false, false);
        __syncthreads();
    }

    // C/D layout: lanes 0-15 -> M=v, lanes 16-31 -> M=v+8; N = lane&15
    const int mh = (lane >> 4) * 8;
    #pragma unroll
    for (int ai = 0; ai < 2; ++ai) {
        #pragma unroll
        for (int bi = 0; bi < 4; ++bi) {
            int col = n0 + wn * 64 + bi * 16 + r;
            if (col < Nt) {
                #pragma unroll
                for (int v = 0; v < 8; ++v) {
                    int m = m0 + wm * 32 + ai * 16 + mh + v;
                    Cout[(size_t)m * Nt + col] = (__bf16)(acc[ai][bi][v] * dinv[m]);
                }
            }
        }
    }
}

// ---------------------------------------------------------------------------
// aggregation (scatter-add of pre-scaled messages)
// ---------------------------------------------------------------------------
__global__ __launch_bounds__(256)
void agg_init_kernel(const __bf16* __restrict__ hws, float* __restrict__ agg, size_t total) {
    size_t idx = (size_t)blockIdx.x * blockDim.x + threadIdx.x;
    if (idx < total) agg[idx] = (float)hws[idx];
}

__global__ __launch_bounds__(256)
void agg_edges_kernel(const int* __restrict__ src, const int* __restrict__ dst,
                      const __bf16* __restrict__ hws, float* __restrict__ agg, int H) {
    int e = blockIdx.x;
    int s = src[e], d = dst[e];
    const __bf16* mrow = hws + (size_t)s * H;
    float* orow = agg + (size_t)d * H;
    for (int j = threadIdx.x; j < H; j += blockDim.x)
        atomicAdd(&orow[j], (float)mrow[j]);
}

// h1_bf[i, j] = j < H ? bf16(leaky(agg[i*H+j]*dinv[i] + b[j])) : 0   (KP padded)
__global__ __launch_bounds__(256)
void bias_act_tobf_kernel(const float* __restrict__ agg, const float* __restrict__ dinv,
                          const float* __restrict__ bias, __bf16* __restrict__ hout,
                          int N, int H, int KP) {
    size_t idx = (size_t)blockIdx.x * blockDim.x + threadIdx.x;
    size_t total = (size_t)N * KP;
    if (idx >= total) return;
    int i = (int)(idx / KP), j = (int)(idx % KP);
    float v = 0.0f;
    if (j < H) {
        v = agg[(size_t)i * H + j] * dinv[i] + bias[j];
        v = LRELU(v);
    }
    hout[idx] = (__bf16)v;
}

// in-place f32: agg[i,j] = leaky(agg[i,j]*dinv[i] + b[j])
__global__ __launch_bounds__(256)
void bias_act_f32_kernel(float* __restrict__ agg, const float* __restrict__ dinv,
                         const float* __restrict__ bias, int N, int H) {
    size_t idx = (size_t)blockIdx.x * blockDim.x + threadIdx.x;
    size_t total = (size_t)N * H;
    if (idx >= total) return;
    int i = (int)(idx / H), j = (int)(idx % H);
    float v = agg[idx] * dinv[i] + bias[j];
    agg[idx] = LRELU(v);
}

// ---------------------------------------------------------------------------
// global max pool (float atomic-max via int/uint trick)
// ---------------------------------------------------------------------------
__device__ inline void atomicMaxF(float* addr, float v) {
    if (v >= 0.0f) atomicMax((int*)addr, __float_as_int(v));
    else           atomicMin((unsigned int*)addr, __float_as_uint(v));
}

__global__ __launch_bounds__(256)
void pool_init_kernel(float* __restrict__ g, size_t total) {
    size_t idx = (size_t)blockIdx.x * blockDim.x + threadIdx.x;
    if (idx < total) g[idx] = -__builtin_inff();
}

__global__ __launch_bounds__(256)
void pool_max_kernel(const float* __restrict__ h, const int* __restrict__ batch,
                     float* __restrict__ g, int H) {
    int i = blockIdx.x;
    int b = batch[i];
    const float* row = h + (size_t)i * H;
    float* grow = g + (size_t)b * H;
    for (int j = threadIdx.x; j < H; j += blockDim.x)
        atomicMaxF(&grow[j], row[j]);
}

// ---------------------------------------------------------------------------
// small dense head: Y[i,j] = leaky(sum_k X[i,k]*W[k,j] + b[j])
// ---------------------------------------------------------------------------
__global__ __launch_bounds__(256)
void mlp_kernel(const float* __restrict__ X, const float* __restrict__ W,
                const float* __restrict__ b, float* __restrict__ Y,
                int M, int K, int Nt) {
    int idx = blockIdx.x * blockDim.x + threadIdx.x;
    if (idx >= M * Nt) return;
    int i = idx / Nt, j = idx % Nt;
    float acc = b[j];
    const float* xr = X + (size_t)i * K;
    for (int k = 0; k < K; ++k) acc += xr[k] * W[(size_t)k * Nt + j];
    Y[idx] = LRELU(acc);
}

// ---------------------------------------------------------------------------
// launcher
// ---------------------------------------------------------------------------
extern "C" void kernel_launch(void* const* d_in, const int* in_sizes, int n_in,
                              void* d_out, int out_size, void* d_ws, size_t ws_size,
                              hipStream_t stream) {
    const int* x_tokens = (const int*)d_in[0];
    const int* ei       = (const int*)d_in[1];
    const int* batch    = (const int*)d_in[2];
    const float* emb    = (const float*)d_in[3];
    const float* W1     = (const float*)d_in[4];
    const float* b1     = (const float*)d_in[5];
    const float* W2     = (const float*)d_in[6];
    const float* b2     = (const float*)d_in[7];
    const float* lw1    = (const float*)d_in[8];
    const float* lb1    = (const float*)d_in[9];
    const float* lw2    = (const float*)d_in[10];
    const float* lb2    = (const float*)d_in[11];
    float* out = (float*)d_out;

    const int N = in_sizes[0];          // 40000
    const int E = in_sizes[1] / 2;      // 80000
    const int G = 128, D = 200, H = 2000;
    const int K1P = 224;                // pad 200 -> 7*32
    const int K2P = 2016;               // pad 2000 -> 63*32
    const int NP  = 2048;               // padded GEMM width (mult of 256)
    const int HB1 = 1000, NOUT = 4;

    const int* src = ei;
    const int* dst = ei + E;

    // workspace layout
    char* base = (char*)d_ws;
    size_t off = 0;
    auto alloc = [&](size_t bytes) { size_t r = off; off += (bytes + 255) & ~(size_t)255; return r; };
    __bf16* h0_bf  = (__bf16*)(base + alloc((size_t)N * K1P * 2));
    __bf16* w1t_bf = (__bf16*)(base + alloc((size_t)NP * K1P * 2));  // transposed
    __bf16* w2t_bf = (__bf16*)(base + alloc((size_t)NP * K2P * 2));  // transposed
    float*  dinv   = (float*) (base + alloc((size_t)N * 4));
    __bf16* hws    = (__bf16*)(base + alloc((size_t)N * H * 2));     // dinv[m]*XW, bf16
    float*  agg    = (float*) (base + alloc((size_t)N * H * 4));
    __bf16* h1_bf  = (__bf16*)(base + alloc((size_t)N * K2P * 2));   // layer-2 GEMM input
    float*  g      = (float*) (base + alloc((size_t)G * H * 4));
    float*  g1     = (float*) (base + alloc((size_t)G * HB1 * 4));

    auto cdiv = [](size_t a, size_t b) { return (unsigned)((a + b - 1) / b); };
    const size_t NH = (size_t)N * H;

    // weights (transposed+padded) / embedding -> bf16
    wcast_padT_kernel<<<cdiv((size_t)NP * K1P, 256), 256, 0, stream>>>(W1, w1t_bf, D, K1P, H, NP);
    wcast_padT_kernel<<<cdiv((size_t)NP * K2P, 256), 256, 0, stream>>>(W2, w2t_bf, H, K2P, H, NP);
    embed_pad_kernel<<<cdiv((size_t)N * K1P, 256), 256, 0, stream>>>(x_tokens, emb, h0_bf, N, D, K1P);

    // degree -> dinv (self loops included)
    deg_init_kernel <<<cdiv(N, 256), 256, 0, stream>>>(dinv, N);
    deg_edges_kernel<<<cdiv(E, 256), 256, 0, stream>>>(dst, dinv, E);
    dinv_kernel     <<<cdiv(N, 256), 256, 0, stream>>>(dinv, N);

    // ---- GCN layer 1 ----
    {
        dim3 grid(NP / TNB, N / TM);
        gemm_bf16_rowscale<<<grid, 256, 0, stream>>>(h0_bf, w1t_bf, dinv, hws, N, K1P, H);
    }
    agg_init_kernel <<<cdiv(NH, 256), 256, 0, stream>>>(hws, agg, NH);
    agg_edges_kernel<<<E, 256, 0, stream>>>(src, dst, hws, agg, H);
    bias_act_tobf_kernel<<<cdiv((size_t)N * K2P, 256), 256, 0, stream>>>(agg, dinv, b1, h1_bf, N, H, K2P);

    // ---- GCN layer 2 ----
    {
        dim3 grid(NP / TNB, N / TM);
        gemm_bf16_rowscale<<<grid, 256, 0, stream>>>(h1_bf, w2t_bf, dinv, hws, N, K2P, H);
    }
    agg_init_kernel <<<cdiv(NH, 256), 256, 0, stream>>>(hws, agg, NH);
    agg_edges_kernel<<<E, 256, 0, stream>>>(src, dst, hws, agg, H);
    bias_act_f32_kernel<<<cdiv(NH, 256), 256, 0, stream>>>(agg, dinv, b2, N, H);

    // ---- global max pool ----
    pool_init_kernel<<<cdiv((size_t)G * H, 256), 256, 0, stream>>>(g, (size_t)G * H);
    pool_max_kernel <<<N, 256, 0, stream>>>(agg, batch, g, H);

    // ---- MLP head ----
    mlp_kernel<<<cdiv((size_t)G * HB1, 256), 256, 0, stream>>>(g, lw1, lb1, g1, G, H, HB1);
    mlp_kernel<<<cdiv((size_t)G * NOUT, 256), 256, 0, stream>>>(g1, lw2, lb2, out, G, HB1, NOUT);
}
